// MagPot_24472723652711
// MI455X (gfx1250) — matrix operand order repaired and checked by reading, not compiled, not run on hardware
//
#include <hip/hip_runtime.h>

typedef __attribute__((ext_vector_type(16))) _Float16     v16h;
typedef __attribute__((ext_vector_type(8)))  float        v8f;
typedef __attribute__((ext_vector_type(4)))  unsigned int v4u;
typedef __attribute__((ext_vector_type(8)))  int          v8i;
typedef __attribute__((ext_vector_type(4)))  int          v4i;

#define N_SPECIES 4
#define BASIS     12
#define N_MAX     8
#define EMBD      16
#define N_HEADS   9
#define ACC_STR   88      // 64 seg + 24 vec floats per atom
#define R_CUTF    6.0f
#define EPSF      1e-9f
#define PI_F      3.14159265358979f

// ---------------------------------------------------------------- utilities
__device__ __forceinline__ float silu_f(float x) {
  // x * v_rcp_f32(1+e^-x): avoids the IEEE div_scale/div_fixup chain
  return x * __builtin_amdgcn_rcpf(1.0f + __expf(-x));
}

__device__ __forceinline__ void atomadd_f32(float* p, float v) {
  unsafeAtomicAdd(p, v);   // lowers to global_atomic_add_f32 on gfx1250
}

// A fragment: 16x32 f16 matrix, row-major in LDS with row stride `stride`.
// ISA layout: lane L (h=L/16, m=L%16): VGPR v holds K pair at
// kb = 2*(v%4) + 16*(v/4) + 8*h.
__device__ __forceinline__ v16h a_frag(const _Float16* X, int stride, int lane) {
  const int h = (lane >> 4) & 1, m = lane & 15;
  v16h a;
#pragma unroll
  for (int v = 0; v < 8; ++v) {
    const int kb = ((v & 3) << 1) + ((v & 4) << 2) + (h << 3);
    a[2 * v]     = X[m * stride + kb];
    a[2 * v + 1] = X[m * stride + kb + 1];
  }
  return a;
}

// B fragment: 32x16 slice of W[k][n] (row stride `stride`), column block nb.
// Lane L (h=L/16, n=nb+L%16): VGPR v holds K = 2*(v%4) + 16*(v/4) + 8*h, +1.
// Rows k >= kmax are zero (K-padding for the 24-row magnetic weights).
__device__ __forceinline__ v16h b_frag(const _Float16* W, int stride, int nb,
                                       int kmax, int lane) {
  const int h = (lane >> 4) & 1, n = nb + (lane & 15);
  v16h b;
#pragma unroll
  for (int v = 0; v < 8; ++v) {
    const int k = ((v & 3) << 1) + ((v & 4) << 2) + (h << 3);
    b[2 * v]     = (k     < kmax) ? W[k * stride + n]       : (_Float16)0.0f;
    b[2 * v + 1] = (k + 1 < kmax) ? W[(k + 1) * stride + n] : (_Float16)0.0f;
  }
  return b;
}

// A fragment for a magnetic head: columns 0..7 = sectors[head], 8..23 = emb,
// 24..31 = 0.  sect: [16][N_HEADS][8] f16, emb: [16][16] f16.
__device__ __forceinline__ v16h a_frag_mag(const _Float16* sect,
                                           const _Float16* emb, int hd,
                                           int lane) {
  const int h = (lane >> 4) & 1, m = lane & 15;
  v16h a;
#pragma unroll
  for (int v = 0; v < 8; ++v) {
    const int kb = ((v & 3) << 1) + ((v & 4) << 2) + (h << 3);
#pragma unroll
    for (int t = 0; t < 2; ++t) {
      const int k = kb + t;
      _Float16 val;
      if (k < 8)        val = sect[(m * N_HEADS + hd) * 8 + k];
      else if (k < 24)  val = emb[m * 16 + (k - 8)];
      else              val = (_Float16)0.0f;
      a[2 * v + t] = val;
    }
  }
  return a;
}

// ---------------------------------------------------------------- kernel 0
__global__ void magpot_zero_kernel(float* acc, float* out, long n_acc,
                                   int n_out) {
  const long stride = (long)gridDim.x * blockDim.x;
  long t = (long)blockIdx.x * blockDim.x + threadIdx.x;
  for (long i = t; i < n_acc; i += stride) acc[i] = 0.0f;
  if (t < n_out) out[t] = 0.0f;
}

// ---------------------------------------------------------------- kernel 1
// One thread per edge: radial basis + magnetic invariants, scatter-add into
// per-atom accumulator acc[atom*88 + ...]. L2-resident fp32 atomics.
// The 6 KB Chebyshev table is staged into LDS by the Tensor Data Mover.
__global__ void __launch_bounds__(256)
magpot_edge_kernel(const float* __restrict__ pos,
                   const float* __restrict__ mom,
                   const int*   __restrict__ species,
                   const int*   __restrict__ i_idx,
                   const int*   __restrict__ j_idx,
                   const float* __restrict__ cheb,  // [4][4][8][12]
                   float* __restrict__ acc, int n_edges) {
  __shared__ float sc[N_SPECIES * N_SPECIES * N_MAX * BASIS];  // 1536 f32

#if __has_builtin(__builtin_amdgcn_tensor_load_to_lds)
  if (threadIdx.x < 32) {  // one wave issues the DMA (TDM ignores EXEC)
    const unsigned long long ga = (unsigned long long)(size_t)cheb;
    const unsigned int lds_off = (unsigned int)(size_t)(void*)&sc[0];
    const unsigned int nelem = N_SPECIES * N_SPECIES * N_MAX * BASIS;  // 1536

    v4u g0;
    g0[0] = 1u;                                    // count=1, user descriptor
    g0[1] = lds_off;                               // lds_addr
    g0[2] = (unsigned int)(ga & 0xFFFFFFFFu);      // global_addr[31:0]
    g0[3] = (unsigned int)((ga >> 32) & 0x01FFFFFFu) | (2u << 30);  // type=2

    v8i g1;
    g1[0] = (int)(2u << 16);                       // data_size = 4 bytes
    g1[1] = (int)((nelem & 0xFFFFu) << 16);        // tensor_dim0[15:0]
    g1[2] = (int)(nelem >> 16);                    // tensor_dim0[31:16]
    g1[3] = (int)((nelem & 0xFFFFu) << 16);        // tile_dim0 = 1536
    g1[4] = 0;                                     // tile_dim1/2 unused
    g1[5] = (int)nelem;                            // tensor_dim0_stride lo
    g1[6] = 0;
    g1[7] = 0;

    v4i g2 = {0, 0, 0, 0};
    v4i g3 = {0, 0, 0, 0};
#if __clang_major__ >= 23
    v8i g4 = {0, 0, 0, 0, 0, 0, 0, 0};
    __builtin_amdgcn_tensor_load_to_lds(g0, g1, g2, g3, g4, 0);
#else
    __builtin_amdgcn_tensor_load_to_lds(g0, g1, g2, g3, 0);
#endif
    __builtin_amdgcn_s_wait_tensorcnt(0);
  }
#else
  for (int t = threadIdx.x; t < N_SPECIES * N_SPECIES * N_MAX * BASIS;
       t += blockDim.x)
    sc[t] = cheb[t];
#endif
  __syncthreads();

  const int e = blockIdx.x * blockDim.x + threadIdx.x;
  if (e >= n_edges) return;

  // stream-ahead prefetch of the (linear) index arrays
  __builtin_prefetch(&i_idx[e + 8192], 0, 1);
  __builtin_prefetch(&j_idx[e + 8192], 0, 1);

  const int i = i_idx[e], j = j_idx[e];
  const float pix = pos[i * 3], piy = pos[i * 3 + 1], piz = pos[i * 3 + 2];
  const float pjx = pos[j * 3], pjy = pos[j * 3 + 1], pjz = pos[j * 3 + 2];
  const float rx = pjx - pix, ry = pjy - piy, rz = pjz - piz;
  const float dist = sqrtf(rx * rx + ry * ry + rz * rz + EPSF);
  const float inv_d = __builtin_amdgcn_rcpf(dist);
  const float hx = rx * inv_d, hy = ry * inv_d, hz = rz * inv_d;

  float x = 2.0f * dist / R_CUTF - 1.0f;
  x = fminf(1.0f, fmaxf(-1.0f, x));
  float T[BASIS];
  T[0] = 1.0f; T[1] = x;
#pragma unroll
  for (int k = 2; k < BASIS; ++k) T[k] = 2.0f * x * T[k - 1] - T[k - 2];
  const float fcut =
      (dist < R_CUTF) ? 0.5f * (__cosf(PI_F * dist / R_CUTF) + 1.0f) : 0.0f;

  const int si = species[i], sj = species[j];
  const float* c = &sc[(si * N_SPECIES + sj) * N_MAX * BASIS];
  float phi[N_MAX];
#pragma unroll
  for (int n = 0; n < N_MAX; ++n) {
    float s = 0.0f;
#pragma unroll
    for (int k = 0; k < BASIS; ++k) s = fmaf(c[n * BASIS + k], T[k], s);
    phi[n] = s * fcut;
  }

  // magnetic invariants
  const float mix = mom[i * 3], miy = mom[i * 3 + 1], miz = mom[i * 3 + 2];
  const float mjx = mom[j * 3], mjy = mom[j * 3 + 1], mjz = mom[j * 3 + 2];
  const float mni = sqrtf(mix * mix + miy * miy + miz * miz + EPSF);
  const float mnj = sqrtf(mjx * mjx + mjy * mjy + mjz * mjz + EPSF);
  const float ii = __builtin_amdgcn_rcpf(mni);
  const float ij = __builtin_amdgcn_rcpf(mnj);
  const float hix = mix * ii, hiy = miy * ii, hiz = miz * ii;
  const float hjx = mjx * ij, hjy = mjy * ij, hjz = mjz * ij;
  const float d_mimj = mix * mjx + miy * mjy + miz * mjz;
  const float d_ir = hix * hx + hiy * hy + hiz * hz;
  const float d_jr = hjx * hx + hjy * hy + hjz * hz;
  const float cx = miy * mjz - miz * mjy;
  const float cy = miz * mjx - mix * mjz;
  const float cz = mix * mjy - miy * mjx;
  const float d_rc = hx * cx + hy * cy + hz * cz;
  const float d_hij = hix * hjx + hiy * hjy + hiz * hjz;

  float wc[8];
  wc[0] = 1.0f;
  wc[1] = d_mimj;
  wc[2] = d_ir * d_ir;
  wc[3] = d_jr * d_jr;
  wc[4] = d_rc;
  wc[5] = mnj;
  wc[6] = d_hij * mnj;
  wc[7] = mnj * mnj;

  float* ai = acc + (size_t)i * ACC_STR;
#pragma unroll
  for (int w = 0; w < 8; ++w)
#pragma unroll
    for (int n = 0; n < N_MAX; ++n)
      atomadd_f32(&ai[w * N_MAX + n], phi[n] * wc[w]);
#pragma unroll
  for (int n = 0; n < N_MAX; ++n) {
    atomadd_f32(&ai[64 + n * 3 + 0], phi[n] * hx);
    atomadd_f32(&ai[64 + n * 3 + 1], phi[n] * hy);
    atomadd_f32(&ai[64 + n * 3 + 2], phi[n] * hz);
  }
}

// ---------------------------------------------------------------- kernel 2
// 64 threads = 2 waves; each wave owns a 16-atom tile and runs both MLPs
// with v_wmma_f32_16x16x32_f16 (f16 A/B in LDS, f32 accumulate).
__global__ void __launch_bounds__(64)
magpot_atom_kernel(const float* __restrict__ acc,
                   const float* __restrict__ mom,
                   const int*   __restrict__ species,
                   const int*   __restrict__ batch,
                   const float* __restrict__ embed,   // [4][16]
                   const float* __restrict__ shift,   // [4]
                   const float* __restrict__ Ws1, const float* __restrict__ bs1,
                   const float* __restrict__ Ws2, const float* __restrict__ bs2,
                   const float* __restrict__ Ws3, const float* __restrict__ bs3,
                   const float* __restrict__ Wm1, const float* __restrict__ bm1,
                   const float* __restrict__ Wm2, const float* __restrict__ bm2,
                   const float* __restrict__ Wm3, const float* __restrict__ bm3,
                   float* __restrict__ out, int n_atoms) {
  __shared__ _Float16 sWs1[32 * 64];                 //  4 KB
  __shared__ _Float16 sWs2[64 * 64];                 //  8 KB
  __shared__ _Float16 sWm1[N_HEADS][24 * 32];        // 13.5 KB
  __shared__ _Float16 sWm2[N_HEADS][32 * 32];        // 18 KB
  __shared__ _Float16 sX[2][16 * 32];                //  2 KB
  __shared__ _Float16 sSect[2][16 * N_HEADS * 8];    //  4.5 KB
  __shared__ _Float16 sEmb[2][16 * 16];              //  1 KB
  __shared__ _Float16 sAct0[2][16 * 64];             //  4 KB
  __shared__ _Float16 sAct1[2][16 * 64];             //  4 KB
  __shared__ float    sRed[2][32];                   //  0.25 KB

  const int tid = threadIdx.x;
  // ---- stage all weights once per block (fp32 -> f16) ----
  for (int i = tid; i < 32 * 64; i += 64) sWs1[i] = (_Float16)Ws1[i];
  for (int i = tid; i < 64 * 64; i += 64) sWs2[i] = (_Float16)Ws2[i];
  {
    _Float16* w1 = &sWm1[0][0];
    for (int i = tid; i < N_HEADS * 24 * 32; i += 64) w1[i] = (_Float16)Wm1[i];
    _Float16* w2 = &sWm2[0][0];
    for (int i = tid; i < N_HEADS * 32 * 32; i += 64) w2[i] = (_Float16)Wm2[i];
  }

  const int wave = tid >> 5, lane = tid & 31;
  const int m = lane & 15, h = (lane >> 4) & 1;
  const int a = (blockIdx.x * 2 + wave) * 16 + m;
  const bool valid = (a < n_atoms);

  _Float16* Xw = sX[wave];
  _Float16* Ew = sEmb[wave];
  _Float16* Sw = sSect[wave];

  // ---- stage per-atom features (lanes 0..15 of each wave) ----
  if (h == 0) {
    float amp = 0.0f;
    if (valid) {
      const float mx = mom[a * 3], my = mom[a * 3 + 1], mz = mom[a * 3 + 2];
      amp = sqrtf(mx * mx + my * my + mz * mz + EPSF);
    }
    const size_t base = (size_t)a * ACC_STR;
    float sg[8][8];
#pragma unroll
    for (int w = 0; w < 8; ++w)
#pragma unroll
      for (int n = 0; n < 8; ++n)
        sg[w][n] = valid ? acc[base + w * 8 + n] : 0.0f;
#pragma unroll
    for (int n = 0; n < 8; ++n) {
      const float v0 = valid ? acc[base + 64 + n * 3 + 0] : 0.0f;
      const float v1 = valid ? acc[base + 64 + n * 3 + 1] : 0.0f;
      const float v2 = valid ? acc[base + 64 + n * 3 + 2] : 0.0f;
      Xw[m * 32 + n]     = (_Float16)sg[0][n];                    // A
      Xw[m * 32 + 8 + n] = (_Float16)(v0 * v0 + v1 * v1 + v2 * v2);
      // sectors: [amp*A, s1, s2, s3, s4, amp*s5, s5, s6, s7]
      Sw[(m * N_HEADS + 0) * 8 + n] = (_Float16)(amp * sg[0][n]);
      Sw[(m * N_HEADS + 1) * 8 + n] = (_Float16)sg[1][n];
      Sw[(m * N_HEADS + 2) * 8 + n] = (_Float16)sg[2][n];
      Sw[(m * N_HEADS + 3) * 8 + n] = (_Float16)sg[3][n];
      Sw[(m * N_HEADS + 4) * 8 + n] = (_Float16)sg[4][n];
      Sw[(m * N_HEADS + 5) * 8 + n] = (_Float16)(amp * sg[5][n]);
      Sw[(m * N_HEADS + 6) * 8 + n] = (_Float16)sg[5][n];
      Sw[(m * N_HEADS + 7) * 8 + n] = (_Float16)sg[6][n];
      Sw[(m * N_HEADS + 8) * 8 + n] = (_Float16)sg[7][n];
    }
    const int sp = valid ? species[a] : 0;
#pragma unroll
    for (int e2 = 0; e2 < EMBD; ++e2) {
      const float ev = embed[sp * EMBD + e2];
      Ew[m * 16 + e2]       = (_Float16)ev;
      Xw[m * 32 + 16 + e2]  = (_Float16)ev;
    }
  }
  __syncthreads();

  const int ncol = lane & 15;

  // ================= structural MLP: 32 -> 64 -> 64 -> 1 =================
  {
    const v16h aX = a_frag(Xw, 32, lane);
#pragma unroll
    for (int nt = 0; nt < 4; ++nt) {
      const float bb = bs1[nt * 16 + ncol];
      v8f c;
#pragma unroll
      for (int v = 0; v < 8; ++v) c[v] = bb;
      const v16h bW = b_frag(sWs1, 64, nt * 16, 32, lane);
      c = __builtin_amdgcn_wmma_f32_16x16x32_f16(false, aX, false, bW,
                                                 (short)0, c, false, false);
      const int n = nt * 16 + ncol;
#pragma unroll
      for (int v = 0; v < 8; ++v)
        sAct0[wave][(v + (h << 3)) * 64 + n] = (_Float16)silu_f(c[v]);
    }
  }
  __syncthreads();
  {
#pragma unroll
    for (int nt = 0; nt < 4; ++nt) {
      const float bb = bs2[nt * 16 + ncol];
      v8f c;
#pragma unroll
      for (int v = 0; v < 8; ++v) c[v] = bb;
#pragma unroll
      for (int kt = 0; kt < 2; ++kt) {
        const v16h aA = a_frag(&sAct0[wave][kt * 32], 64, lane);
        const v16h bW = b_frag(&sWs2[kt * 32 * 64], 64, nt * 16, 32, lane);
        c = __builtin_amdgcn_wmma_f32_16x16x32_f16(false, aA, false, bW,
                                                   (short)0, c, false, false);
      }
      const int n = nt * 16 + ncol;
#pragma unroll
      for (int v = 0; v < 8; ++v)
        sAct1[wave][(v + (h << 3)) * 64 + n] = (_Float16)silu_f(c[v]);
    }
  }
  __syncthreads();
  float part = 0.0f;
#pragma unroll
  for (int k = 0; k < 32; ++k) {
    const int kk = h * 32 + k;
    part = fmaf((float)sAct1[wave][m * 64 + kk], Ws3[kk], part);
  }
  sRed[wave][lane] = part;
  __syncthreads();
  float e_at = 0.0f;
  if (h == 0) e_at = sRed[wave][m] + sRed[wave][m + 16] + bs3[0];

  // ================= magnetic heads: 9 x (24 -> 32 -> 32 -> 1) ===========
  for (int hd = 0; hd < N_HEADS; ++hd) {
    __syncthreads();
    const v16h aXh = a_frag_mag(Sw, Ew, hd, lane);
#pragma unroll
    for (int nt = 0; nt < 2; ++nt) {
      const float bb = bm1[hd * 32 + nt * 16 + ncol];
      v8f c;
#pragma unroll
      for (int v = 0; v < 8; ++v) c[v] = bb;
      const v16h bW = b_frag(sWm1[hd], 32, nt * 16, 24, lane);
      c = __builtin_amdgcn_wmma_f32_16x16x32_f16(false, aXh, false, bW,
                                                 (short)0, c, false, false);
      const int n = nt * 16 + ncol;
#pragma unroll
      for (int v = 0; v < 8; ++v)
        sAct0[wave][(v + (h << 3)) * 32 + n] = (_Float16)silu_f(c[v]);
    }
    __syncthreads();
    {
      const v16h aA = a_frag(sAct0[wave], 32, lane);
#pragma unroll
      for (int nt = 0; nt < 2; ++nt) {
        const float bb = bm2[hd * 32 + nt * 16 + ncol];
        v8f c;
#pragma unroll
        for (int v = 0; v < 8; ++v) c[v] = bb;
        const v16h bW = b_frag(sWm2[hd], 32, nt * 16, 32, lane);
        c = __builtin_amdgcn_wmma_f32_16x16x32_f16(false, aA, false, bW,
                                                   (short)0, c, false, false);
        const int n = nt * 16 + ncol;
#pragma unroll
        for (int v = 0; v < 8; ++v)
          sAct1[wave][(v + (h << 3)) * 32 + n] = (_Float16)silu_f(c[v]);
      }
    }
    __syncthreads();
    float p = 0.0f;
#pragma unroll
    for (int k = 0; k < 16; ++k) {
      const int kk = h * 16 + k;
      p = fmaf((float)sAct1[wave][m * 32 + kk], Wm3[hd * 32 + kk], p);
    }
    sRed[wave][lane] = p;
    __syncthreads();
    if (h == 0) e_at += sRed[wave][m] + sRed[wave][m + 16] + bm3[hd];
  }

  // ---- per-atom energy -> structure bins ----
  if (h == 0 && valid) {
    e_at += shift[species[a]];
    atomadd_f32(&out[batch[a]], e_at);
  }
}

// ---------------------------------------------------------------- launcher
extern "C" void kernel_launch(void* const* d_in, const int* in_sizes, int n_in,
                              void* d_out, int out_size, void* d_ws,
                              size_t ws_size, hipStream_t stream) {
  const float* positions = (const float*)d_in[0];
  const float* moments   = (const float*)d_in[1];
  const int*   species   = (const int*)d_in[2];
  const int*   i_idx     = (const int*)d_in[3];
  const int*   j_idx     = (const int*)d_in[4];
  const int*   batch     = (const int*)d_in[5];
  const float* cheb      = (const float*)d_in[6];
  const float* embed     = (const float*)d_in[7];
  const float* shift     = (const float*)d_in[8];
  const float* Ws1 = (const float*)d_in[9];
  const float* bs1 = (const float*)d_in[10];
  const float* Ws2 = (const float*)d_in[11];
  const float* bs2 = (const float*)d_in[12];
  const float* Ws3 = (const float*)d_in[13];
  const float* bs3 = (const float*)d_in[14];
  const float* Wm1 = (const float*)d_in[15];
  const float* bm1 = (const float*)d_in[16];
  const float* Wm2 = (const float*)d_in[17];
  const float* bm2 = (const float*)d_in[18];
  const float* Wm3 = (const float*)d_in[19];
  const float* bm3 = (const float*)d_in[20];
  float* out = (float*)d_out;
  float* acc = (float*)d_ws;  // n_atoms * 88 floats (~35 MB, L2-resident)

  const int  n_atoms = in_sizes[2];
  const int  n_edges = in_sizes[3];
  const long n_acc   = (long)n_atoms * ACC_STR;

  magpot_zero_kernel<<<1024, 256, 0, stream>>>(acc, out, n_acc, out_size);

  magpot_edge_kernel<<<(n_edges + 255) / 256, 256, 0, stream>>>(
      positions, moments, species, i_idx, j_idx, cheb, acc, n_edges);

  const int n_tiles  = (n_atoms + 15) / 16;
  const int n_blocks = (n_tiles + 1) / 2;
  magpot_atom_kernel<<<n_blocks, 64, 0, stream>>>(
      acc, moments, species, batch, embed, shift, Ws1, bs1, Ws2, bs2, Ws3, bs3,
      Wm1, bm1, Wm2, bm2, Wm3, bm3, out, n_atoms);
}
